// EdgePrediction_85366769975821
// MI455X (gfx1250) — matrix operand order repaired and checked by reading, not compiled, not run on hardware
//
#include <hip/hip_runtime.h>
#include <hip/hip_bf16.h>

typedef __attribute__((ext_vector_type(16))) _Float16 v16h;
typedef __attribute__((ext_vector_type(8)))  _Float16 v8h;
typedef __attribute__((ext_vector_type(8)))  float    v8f;

__device__ __forceinline__ float gelu_f(float x){
  return 0.5f * x * (1.0f + erff(x * 0.70710678118654752f));
}
// Monotonic order-preserving float->uint key for atomicMax-based segment max.
__device__ __forceinline__ unsigned f32_ord_key(float f){
  unsigned u = __float_as_uint(f);
  return (u & 0x80000000u) ? ~u : (u | 0x80000000u);
}
__device__ __forceinline__ float f32_ord_unkey(unsigned u){
  unsigned b = (u & 0x80000000u) ? (u & 0x7FFFFFFFu) : ~u;
  return __uint_as_float(b);
}
__device__ __forceinline__ v8h pack8(float4 a, float4 b){
  v8h r;
  r[0]=(_Float16)a.x; r[1]=(_Float16)a.y; r[2]=(_Float16)a.z; r[3]=(_Float16)a.w;
  r[4]=(_Float16)b.x; r[5]=(_Float16)b.y; r[6]=(_Float16)b.z; r[7]=(_Float16)b.w;
  return r;
}

// ---------------------------------------------------------------------------
// WMMA GEMM workhorse: C[M,128] = act(A[M,K] @ W[K,128] + bias),  K = KT*32
//   A row-major (lda==K), W row-major (ldw==128; row-offset slices OK).
//   8 waves / 256-thread block; each wave owns a 16-row strip x 128 cols.
//   sA: A tile in CDNA5 16-bit A-fragment order -> per-wave v16h is one
//       contiguous 32B LDS load.  Staging = 4x float4 global loads (two
//       contiguous 8-float runs per lane: k = base+0..7 / base+16..23,
//       base=(lane&16)?8:0) packed to 2x ds_store_b128, single row<M branch.
//   sW: column-major f16 image sW[col][k].  Staging is fully coalesced
//       (adjacent threads = adjacent cols of a W row) with packed b128
//       stores; the B-fragment (half h -> K=k0+h+(lane>=16?8:0)+(h>=8?8:0), fixed col)
//       is then two v8h LDS loads joined into one v16h.
// ---------------------------------------------------------------------------
template<int KT>
__global__ __launch_bounds__(256)
void k_gemm128_wmma(const float* __restrict__ A, int M,
                    const float* __restrict__ W,
                    const float* __restrict__ bias,
                    float* __restrict__ C, int act)
{
  constexpr int K = KT * 32;
  __shared__ __attribute__((aligned(32))) _Float16 sW[128 * K]; // [col][k]
  __shared__ __attribute__((aligned(32))) _Float16 sA[8*32*16]; // [r][l][h]
  const int tid  = threadIdx.x;
  const int lane = tid & 31;
  const int wave = tid >> 5;
  const int brow = blockIdx.x * 128;

  // ---- stage W (coalesced): thread -> (col, 16-k chunk) ----
  {
    const int col  = tid & 127;
    const int kh   = (tid >> 7) << 4;      // 0 or 16
    #pragma unroll
    for (int p = 0; p < KT; ++p){
      const int kbase = p*32 + kh;
      const float* wp = W + (size_t)kbase*128 + col;
      float4 w0, w1, w2, w3;
      w0.x = wp[0*128];  w0.y = wp[1*128];  w0.z = wp[2*128];  w0.w = wp[3*128];
      w1.x = wp[4*128];  w1.y = wp[5*128];  w1.z = wp[6*128];  w1.w = wp[7*128];
      w2.x = wp[8*128];  w2.y = wp[9*128];  w2.z = wp[10*128]; w2.w = wp[11*128];
      w3.x = wp[12*128]; w3.y = wp[13*128]; w3.z = wp[14*128]; w3.w = wp[15*128];
      *(v8h*)&sW[(size_t)col*K + kbase]     = pack8(w0, w1);
      *(v8h*)&sW[(size_t)col*K + kbase + 8] = pack8(w2, w3);
    }
  }
  __syncthreads();

  v8f acc[8];
  #pragma unroll
  for (int j = 0; j < 8; ++j) acc[j] = (v8f)0.0f;

  // per-thread A staging geometry (fixed across k-steps)
  const int ar   = tid >> 5;
  const int al   = tid & 31;
  const int arow = brow + ar*16 + (al & 15);
  const int abase = (al & 16) ? 8 : 0;
  const bool arow_ok = (arow < M);
  const float* ap = A + (size_t)arow * K + abase;

  #pragma unroll
  for (int t = 0; t < KT; ++t){
    {
      float4 a0 = {0,0,0,0}, a1 = {0,0,0,0}, a2 = {0,0,0,0}, a3 = {0,0,0,0};
      if (arow_ok){
        const float* p = ap + t*32;
        a0 = *(const float4*)(p);
        a1 = *(const float4*)(p + 4);
        a2 = *(const float4*)(p + 16);
        a3 = *(const float4*)(p + 20);
        if (t + 1 < KT) __builtin_prefetch(p + 32, 0, 0);
      }
      *(v8h*)&sA[(tid << 4)]     = pack8(a0, a1);   // halves 0..7  : k=base+0..7
      *(v8h*)&sA[(tid << 4) + 8] = pack8(a2, a3);   // halves 8..15 : k=base+16..23
    }
    __syncthreads();
    v16h afrag = *(const v16h*)&sA[(wave*32 + lane) << 4];
    const int boff = t*32 + ((lane & 16) ? 8 : 0);
    #pragma unroll
    for (int j = 0; j < 8; ++j){
      const int col = j*16 + (lane & 15);
      v8h blo = *(const v8h*)&sW[(size_t)col*K + boff];
      v8h bhi = *(const v8h*)&sW[(size_t)col*K + boff + 16];
      v16h bfrag = __builtin_shufflevector(blo, bhi, 0,1,2,3,4,5,6,7,8,9,10,11,12,13,14,15);
      acc[j] = __builtin_amdgcn_wmma_f32_16x16x32_f16(
                 false, afrag, false, bfrag, (short)0, acc[j], false, false);
    }
    __syncthreads();
  }

  // C/D layout: VGPR i -> M = i (lanes 0-15) or i+8 (lanes 16-31); N = lane&15
  const int rbase = brow + wave*16 + ((lane & 16) ? 8 : 0);
  if (brow + 128 <= M){
    #pragma unroll
    for (int j = 0; j < 8; ++j){
      int col = j*16 + (lane & 15);
      float bv = bias ? bias[col] : 0.0f;
      #pragma unroll
      for (int i = 0; i < 8; ++i){
        float v = acc[j][i] + bv;
        if (act == 1) v = gelu_f(v);
        C[(size_t)(rbase + i)*128 + col] = v;
      }
    }
  } else {
    #pragma unroll
    for (int j = 0; j < 8; ++j){
      int col = j*16 + (lane & 15);
      float bv = bias ? bias[col] : 0.0f;
      #pragma unroll
      for (int i = 0; i < 8; ++i){
        int row = rbase + i;
        if (row < M){
          float v = acc[j][i] + bv;
          if (act == 1) v = gelu_f(v);
          C[(size_t)row*128 + col] = v;
        }
      }
    }
  }
}

// ---------------------------------------------------------------------------
// Graph preprocessing: per-dst edge count + summed edge attrs (for PyG
// fill_value='mean' self-loop attributes).
// ---------------------------------------------------------------------------
__global__ void k_cnt_sumattr(const int* __restrict__ dst,
                              const float* __restrict__ attr,
                              float* __restrict__ cnt, float* __restrict__ sum_attr,
                              int E_)
{
  int e = blockIdx.x*blockDim.x + threadIdx.x;
  if (e >= E_) return;
  int d = dst[e];
  atomicAdd(&cnt[d], 1.0f);
  #pragma unroll
  for (int a = 0; a < 16; ++a)
    atomicAdd(&sum_attr[(size_t)d*16 + a], attr[(size_t)e*16 + a]);
}

// loop_attr[n] = (sum_attr[n]@attr1_w + cnt[n]*attr1_b) / max(cnt[n],1)
__global__ __launch_bounds__(256)
void k_loop_attr(const float* __restrict__ sum_attr, const float* __restrict__ cnt,
                 const float* __restrict__ w1, const float* __restrict__ b1,
                 float* __restrict__ loop_attr, int Nn)
{
  __shared__ float sw[16*128];
  __shared__ float sb[128];
  for (int i = threadIdx.x; i < 2048; i += 256) sw[i] = w1[i];
  if (threadIdx.x < 128) sb[threadIdx.x] = b1[threadIdx.x];
  __syncthreads();
  int lane = threadIdx.x & 31;
  int n = blockIdx.x*8 + (threadIdx.x >> 5);
  if (n >= Nn) return;
  float c  = cnt[n];
  float sa = (lane < 16) ? sum_attr[(size_t)n*16 + lane] : 0.0f;
  float inv = 1.0f / fmaxf(c, 1.0f);
  float acc[4];
  #pragma unroll
  for (int i = 0; i < 4; ++i) acc[i] = c * sb[lane + i*32];
  for (int a = 0; a < 16; ++a){
    float aa = __shfl(sa, a, 32);
    #pragma unroll
    for (int i = 0; i < 4; ++i) acc[i] += aa * sw[a*128 + lane + i*32];
  }
  #pragma unroll
  for (int i = 0; i < 4; ++i) loop_attr[(size_t)n*128 + lane + i*32] = acc[i] * inv;
}

// ---------------------------------------------------------------------------
// GATv2 attention score, wave per row (E edges + N self loops).
// Edge rows fuse eproj = attr[e] @ cWe (+ attr1_b@We) as a 16->128 matvec.
// ---------------------------------------------------------------------------
__global__ __launch_bounds__(256)
void k_att_score(const int* __restrict__ src, const int* __restrict__ dst,
                 int E_, int R,
                 const float* __restrict__ attr,
                 const float* __restrict__ cWe,    // [16,128] = attr1_w @ We
                 const float* __restrict__ attbv,  // [128]    = attr1_b @ We
                 const float* __restrict__ lproj,  // [N,128]  = loop_attr @ We
                 const float* __restrict__ xl, const float* __restrict__ xr,
                 const float* __restrict__ att,
                 float* __restrict__ s_out, unsigned* __restrict__ smaxkey)
{
  __shared__ float sC[16*128];
  __shared__ float satt[128];
  __shared__ float sab[128];
  for (int i = threadIdx.x; i < 2048; i += 256) sC[i] = cWe[i];
  if (threadIdx.x < 128){ satt[threadIdx.x] = att[threadIdx.x]; sab[threadIdx.x] = attbv[threadIdx.x]; }
  __syncthreads();
  int lane = threadIdx.x & 31;
  int row = blockIdx.x*8 + (threadIdx.x >> 5);
  if (row >= R) return;
  int s_i, d_i;
  float ep[4];
  if (row < E_){
    s_i = src[row]; d_i = dst[row];
    float av = (lane < 16) ? attr[(size_t)row*16 + lane] : 0.0f;
    #pragma unroll
    for (int i = 0; i < 4; ++i) ep[i] = sab[lane + i*32];
    for (int a = 0; a < 16; ++a){
      float aa = __shfl(av, a, 32);
      #pragma unroll
      for (int i = 0; i < 4; ++i) ep[i] += aa * sC[a*128 + lane + i*32];
    }
  } else {
    int n = row - E_;
    s_i = n; d_i = n;
    #pragma unroll
    for (int i = 0; i < 4; ++i) ep[i] = lproj[(size_t)n*128 + lane + i*32];
  }
  float p = 0.0f;
  #pragma unroll
  for (int i = 0; i < 4; ++i){
    int f = lane + i*32;
    float m = xl[(size_t)s_i*128 + f] + xr[(size_t)d_i*128 + f] + ep[i];
    float lr = (m > 0.0f) ? m : 0.2f*m;
    p += lr * satt[f];
  }
  for (int o = 16; o; o >>= 1) p += __shfl_xor(p, o, 32);
  if (lane == 0){
    s_out[row] = p;
    atomicMax(&smaxkey[d_i], f32_ord_key(p));
  }
}

__global__ void k_att_exp(const int* __restrict__ dst, int E_, int R,
                          const float* __restrict__ s_out,
                          const unsigned* __restrict__ smaxkey,
                          float* __restrict__ exbuf, float* __restrict__ den)
{
  int row = blockIdx.x*blockDim.x + threadIdx.x;
  if (row >= R) return;
  int d_i = (row < E_) ? dst[row] : (row - E_);
  float ex = expf(s_out[row] - f32_ord_unkey(smaxkey[d_i]));
  exbuf[row] = ex;
  atomicAdd(&den[d_i], ex);
}

__global__ __launch_bounds__(256)
void k_att_aggr(const int* __restrict__ src, const int* __restrict__ dst,
                int E_, int R,
                const float* __restrict__ exbuf, const float* __restrict__ den,
                const float* __restrict__ xl, float* __restrict__ outacc)
{
  int lane = threadIdx.x & 31;
  int row = blockIdx.x*8 + (threadIdx.x >> 5);
  if (row >= R) return;
  int s_i, d_i;
  if (row < E_){ s_i = src[row]; d_i = dst[row]; }
  else         { s_i = d_i = row - E_; }
  float alpha = exbuf[row] / den[d_i];
  #pragma unroll
  for (int i = 0; i < 4; ++i){
    int f = lane + i*32;
    atomicAdd(&outacc[(size_t)d_i*128 + f], alpha * xl[(size_t)s_i*128 + f]);
  }
}

// h = gelu(outacc + conv_bias) + pre
__global__ void k_node_update(const float* __restrict__ outacc,
                              const float* __restrict__ bias,
                              const float* __restrict__ pre,
                              float* __restrict__ x, int total)
{
  int i = blockIdx.x*blockDim.x + threadIdx.x;
  if (i >= total) return;
  x[i] = gelu_f(outacc[i] + bias[i & 127]) + pre[i];
}

// ---------------------------------------------------------------------------
// mlp0 fusion epilogue: xf[e] already holds edge_embed@W0a; add
// attr[e]@Wc2 + hP1[src]+hQ1[dst]+nP2[src]+nQ2[dst] + c0, apply gelu in place.
// ---------------------------------------------------------------------------
__global__ __launch_bounds__(256)
void k_edge_fuse(const int* __restrict__ src, const int* __restrict__ dst,
                 const float* __restrict__ attr,
                 const float* __restrict__ Wc2, const float* __restrict__ c0,
                 const float* __restrict__ hP1, const float* __restrict__ hQ1,
                 const float* __restrict__ nP2, const float* __restrict__ nQ2,
                 float* __restrict__ xf, int E_)
{
  __shared__ float sw[16*128];
  __shared__ float sc[128];
  for (int i = threadIdx.x; i < 2048; i += 256) sw[i] = Wc2[i];
  if (threadIdx.x < 128) sc[threadIdx.x] = c0[threadIdx.x];
  __syncthreads();
  int lane = threadIdx.x & 31;
  int e = blockIdx.x*8 + (threadIdx.x >> 5);
  if (e >= E_) return;
  int s_i = src[e], d_i = dst[e];
  float av = (lane < 16) ? attr[(size_t)e*16 + lane] : 0.0f;
  float z[4];
  #pragma unroll
  for (int i = 0; i < 4; ++i){
    int f = lane + i*32;
    z[i] = xf[(size_t)e*128 + f] + sc[f]
         + hP1[(size_t)s_i*128 + f] + hQ1[(size_t)d_i*128 + f]
         + nP2[(size_t)s_i*128 + f] + nQ2[(size_t)d_i*128 + f];
  }
  for (int a = 0; a < 16; ++a){
    float aa = __shfl(av, a, 32);
    #pragma unroll
    for (int i = 0; i < 4; ++i) z[i] += aa * sw[a*128 + lane + i*32];
  }
  #pragma unroll
  for (int i = 0; i < 4; ++i) xf[(size_t)e*128 + lane + i*32] = gelu_f(z[i]);
}

// out[e] = dot(xf[e], w3) + b3
__global__ __launch_bounds__(256)
void k_mlp3(const float* __restrict__ xf, const float* __restrict__ w3,
            const float* __restrict__ b3, float* __restrict__ out, int E_)
{
  __shared__ float sw[128];
  if (threadIdx.x < 128) sw[threadIdx.x] = w3[threadIdx.x];
  __syncthreads();
  int lane = threadIdx.x & 31;
  int e = blockIdx.x*8 + (threadIdx.x >> 5);
  if (e >= E_) return;
  float p = 0.0f;
  #pragma unroll
  for (int i = 0; i < 4; ++i) p += xf[(size_t)e*128 + lane + i*32] * sw[lane + i*32];
  for (int o = 16; o; o >>= 1) p += __shfl_xor(p, o, 32);
  if (lane == 0) out[e] = p + b3[0];
}

// out[col] (+)= v[K] @ W[K,128]   (single block, 128 threads; v = vbase[+idx*128])
__global__ void k_vecmat(const float* __restrict__ vbase, const int* __restrict__ idx,
                         const float* __restrict__ W, int K,
                         float* __restrict__ out, int accumulate)
{
  int col = threadIdx.x;
  const float* v = vbase + (idx ? (size_t)idx[0] * 128 : 0);
  float s = accumulate ? out[col] : 0.0f;
  for (int k = 0; k < K; ++k) s += v[k] * W[k*128 + col];
  out[col] = s;
}

__global__ void k_fill_f32(float* p, int n, float v){
  int i = blockIdx.x*blockDim.x + threadIdx.x;
  if (i < n) p[i] = v;
}
__global__ void k_fill_u32(unsigned* p, int n, unsigned v){
  int i = blockIdx.x*blockDim.x + threadIdx.x;
  if (i < n) p[i] = v;
}

// ---------------------------------------------------------------------------
extern "C" void kernel_launch(void* const* d_in, const int* in_sizes, int n_in,
                              void* d_out, int out_size, void* d_ws, size_t ws_size,
                              hipStream_t stream) {
  const float* x_rec      = (const float*)d_in[0];
  const int*   edge_index = (const int*)  d_in[1];
  const float* attr       = (const float*)d_in[2];
  const int*   cur_t      = (const int*)  d_in[3];
  const int*   cur_w      = (const int*)  d_in[4];
  const float* node_embed = (const float*)d_in[5];
  const float* edge_embed = (const float*)d_in[6];
  const float* time_embed = (const float*)d_in[7];
  const float* week_embed = (const float*)d_in[8];
  const float* node_lin_w = (const float*)d_in[9];
  const float* node_lin_b = (const float*)d_in[10];
  const float* attr1_w    = (const float*)d_in[11];
  const float* attr1_b    = (const float*)d_in[12];
  const float* attr2_w    = (const float*)d_in[13];
  const float* attr2_b    = (const float*)d_in[14];
  const float* c_Wl[2]   = {(const float*)d_in[15], (const float*)d_in[22]};
  const float* c_bl[2]   = {(const float*)d_in[16], (const float*)d_in[23]};
  const float* c_Wr[2]   = {(const float*)d_in[17], (const float*)d_in[24]};
  const float* c_br[2]   = {(const float*)d_in[18], (const float*)d_in[25]};
  const float* c_We[2]   = {(const float*)d_in[19], (const float*)d_in[26]};
  const float* c_att[2]  = {(const float*)d_in[20], (const float*)d_in[27]};
  const float* c_bias[2] = {(const float*)d_in[21], (const float*)d_in[28]};
  const float* gnn1_w = (const float*)d_in[29];
  const float* gnn1_b = (const float*)d_in[30];
  const float* gnn2_w = (const float*)d_in[31];
  const float* gnn2_b = (const float*)d_in[32];
  const float* mlp0_w = (const float*)d_in[33];
  const float* mlp0_b = (const float*)d_in[34];
  const float* mlp1_w = (const float*)d_in[35];
  const float* mlp1_b = (const float*)d_in[36];
  const float* mlp2_w = (const float*)d_in[37];
  const float* mlp2_b = (const float*)d_in[38];
  const float* mlp3_w = (const float*)d_in[39];
  const float* mlp3_b = (const float*)d_in[40];

  const int Nn = in_sizes[5] / 128;
  const int Ee = in_sizes[1] / 2;
  const int IN_ = in_sizes[0] / Nn;
  const int R  = Ee + Nn;
  const int* srcI = edge_index;
  const int* dstI = edge_index + Ee;

  // Bump allocator over workspace
  char* wsp = (char*)d_ws; size_t off = 0;
  auto alloc = [&](size_t bytes) -> float* {
    float* p = (float*)(wsp + off);
    off = (off + bytes + 255) & ~(size_t)255;
    return p;
  };
  float* cnt       = alloc((size_t)Nn * 4);
  float* sum_attr  = alloc((size_t)Nn * 16 * 4);
  float* loop_attr = alloc((size_t)Nn * 128 * 4);
  float* P1  = alloc(128*128*4);
  float* Q1  = alloc(128*128*4);
  float* P2  = alloc(128*128*4);
  float* Q2  = alloc(128*128*4);
  float* Wc2 = alloc(16*128*4);
  float* cWe = alloc(6 * 16*128*4);     // [branch*3+layer]
  float* attbv = alloc(6 * 128*4);
  float* c0  = alloc(128*4);
  float* xlb = alloc((size_t)Nn * 128 * 4);
  float* xrb = alloc((size_t)Nn * 128 * 4);
  float* lpj = alloc((size_t)Nn * 128 * 4);
  float* outacc = alloc((size_t)Nn * 128 * 4);
  float* s_out = alloc((size_t)R * 4);
  float* exbuf = alloc((size_t)R * 4);
  float* den   = alloc((size_t)Nn * 4);
  unsigned* smaxkey = (unsigned*)alloc((size_t)Nn * 4);
  float* hbuf  = alloc((size_t)Nn * 128 * 4);
  float* nfbuf = alloc((size_t)Nn * 128 * 4);
  float* pre2  = alloc((size_t)Nn * 128 * 4);
  float* xfA = alloc((size_t)Ee * 128 * 4);
  float* xfB = alloc((size_t)Ee * 128 * 4);
  // after GAT layers finish, reuse attention scratch for the head composites
  float* hP1 = xlb; float* hQ1 = xrb; float* nP2 = lpj; float* nQ2 = outacc;

  auto gemm = [&](const float* A, int M, int K, const float* W,
                  const float* bias, float* C, int act){
    dim3 g((M + 127)/128);
    if (K == 128)
      k_gemm128_wmma<4><<<g, 256, 0, stream>>>(A, M, W, bias, C, act);
    else
      k_gemm128_wmma<2><<<g, 256, 0, stream>>>(A, M, W, bias, C, act);
  };
  auto fillf = [&](float* p, int n, float v){
    k_fill_f32<<<dim3((n + 255)/256), 256, 0, stream>>>(p, n, v);
  };

  // ---- preprocessing: self-loop mean attributes ----
  fillf(cnt, Nn, 0.0f);
  fillf(sum_attr, Nn*16, 0.0f);
  k_cnt_sumattr<<<dim3((Ee + 255)/256), 256, 0, stream>>>(dstI, attr, cnt, sum_attr, Ee);
  k_loop_attr<<<dim3((Nn + 7)/8), 256, 0, stream>>>(sum_attr, cnt, attr1_w, attr1_b, loop_attr, Nn);

  // ---- composite weights (fold gnn heads + attr projections into mlp0) ----
  gemm(gnn1_w,           128, 128, mlp0_w + 256*128, nullptr, P1, 0);
  gemm(gnn1_w + 128*128, 128, 128, mlp0_w + 256*128, nullptr, Q1, 0);
  gemm(gnn2_w,           128, 128, mlp0_w + 384*128, nullptr, P2, 0);
  gemm(gnn2_w + 128*128, 128, 128, mlp0_w + 384*128, nullptr, Q2, 0);
  gemm(attr2_w, 16, 128, mlp0_w + 128*128, nullptr, Wc2, 0);
  for (int b = 0; b < 2; ++b)
    for (int l = 0; l < 3; ++l){
      gemm(attr1_w, 16, 128, c_We[b] + (size_t)l*128*128, nullptr, cWe + (b*3+l)*16*128, 0);
      k_vecmat<<<1, 128, 0, stream>>>(attr1_b, nullptr, c_We[b] + (size_t)l*128*128, 128,
                                      attbv + (b*3+l)*128, 0);
    }
  // c0 = mlp0_b + gnn1_b@W0c + gnn2_b@W0d + attr2_b@W0b + t@W0e + w@W0f
  hipMemcpyAsync(c0, mlp0_b, 128*4, hipMemcpyDeviceToDevice, stream);
  k_vecmat<<<1, 128, 0, stream>>>(gnn1_b, nullptr, mlp0_w + 256*128, 128, c0, 1);
  k_vecmat<<<1, 128, 0, stream>>>(gnn2_b, nullptr, mlp0_w + 384*128, 128, c0, 1);
  k_vecmat<<<1, 128, 0, stream>>>(attr2_b, nullptr, mlp0_w + 128*128, 128, c0, 1);
  k_vecmat<<<1, 128, 0, stream>>>(time_embed, cur_t, mlp0_w + 512*128, 128, c0, 1);
  k_vecmat<<<1, 128, 0, stream>>>(week_embed, cur_w, mlp0_w + 640*128, 128, c0, 1);

  // ---- branch inputs ----
  hipMemcpyAsync(hbuf, node_embed, (size_t)Nn*128*4, hipMemcpyDeviceToDevice, stream);
  gemm(x_rec, Nn, IN_, node_lin_w, node_lin_b, nfbuf, 1);  // nf0 = gelu(...)
  hipMemcpyAsync(pre2, nfbuf, (size_t)Nn*128*4, hipMemcpyDeviceToDevice, stream);

  // ---- 2 branches x 3 GATv2 layers ----
  for (int b = 0; b < 2; ++b){
    float* x = b ? nfbuf : hbuf;
    const float* pre = b ? (const float*)pre2 : node_embed;
    for (int l = 0; l < 3; ++l){
      const size_t wo = (size_t)l * 128 * 128;
      gemm(x, Nn, 128, c_Wl[b] + wo, c_bl[b] + l*128, xlb, 0);
      gemm(x, Nn, 128, c_Wr[b] + wo, c_br[b] + l*128, xrb, 0);
      gemm(loop_attr, Nn, 128, c_We[b] + wo, nullptr, lpj, 0);
      k_fill_u32<<<dim3((Nn + 255)/256), 256, 0, stream>>>(smaxkey, Nn, 0u);
      fillf(den, Nn, 0.0f);
      fillf(outacc, Nn*128, 0.0f);
      k_att_score<<<dim3((R + 7)/8), 256, 0, stream>>>(
          srcI, dstI, Ee, R, attr, cWe + (b*3+l)*16*128, attbv + (b*3+l)*128,
          lpj, xlb, xrb, c_att[b] + l*128, s_out, smaxkey);
      k_att_exp<<<dim3((R + 255)/256), 256, 0, stream>>>(
          dstI, Ee, R, s_out, smaxkey, exbuf, den);
      k_att_aggr<<<dim3((R + 7)/8), 256, 0, stream>>>(
          srcI, dstI, Ee, R, exbuf, den, xlb, outacc);
      k_node_update<<<dim3((Nn*128 + 255)/256), 256, 0, stream>>>(
          outacc, c_bias[b] + l*128, pre, x, Nn*128);
    }
  }

  // ---- node-level head composites (reuse attention scratch buffers) ----
  gemm(hbuf,  Nn, 128, P1, nullptr, hP1, 0);
  gemm(hbuf,  Nn, 128, Q1, nullptr, hQ1, 0);
  gemm(nfbuf, Nn, 128, P2, nullptr, nP2, 0);
  gemm(nfbuf, Nn, 128, Q2, nullptr, nQ2, 0);

  // ---- edge MLP ----
  gemm(edge_embed, Ee, 128, mlp0_w, nullptr, xfA, 0);           // edge_embed @ W0a
  k_edge_fuse<<<dim3((Ee + 7)/8), 256, 0, stream>>>(
      srcI, dstI, attr, Wc2, c0, hP1, hQ1, nP2, nQ2, xfA, Ee);  // + gathers, gelu
  gemm(xfA, Ee, 128, mlp1_w, mlp1_b, xfB, 1);
  gemm(xfB, Ee, 128, mlp2_w, mlp2_b, xfA, 1);
  k_mlp3<<<dim3((Ee + 7)/8), 256, 0, stream>>>(xfA, mlp3_w, mlp3_b, (float*)d_out, Ee);
}
